// MoGEBlock_83124797046954
// MI455X (gfx1250) — matrix operand order repaired and checked by reading, not compiled, not run on hardware
//
#include <hip/hip_runtime.h>

// ---------------- problem constants ----------------
#define NTOK 4096
#define DMODEL 1024
#define NEXP 8
#define DFF 4096
#define KTOP 2
#define SLOTS (NTOK * KTOP)         // 8192 routed token-slots
#define TM 64                        // tokens per block tile
#define TN 128                       // output cols per block tile
#define KS 32                        // K step (bf16 WMMA K)
#define MAX_TILES (SLOTS / TM + NEXP) // 136, worst-case expert tiles

typedef __attribute__((ext_vector_type(16))) __bf16 bf16x16;
typedef __attribute__((ext_vector_type(8)))  float  f32x8;

union BFrag { bf16x16 v; uint4 u[2]; };

__device__ __forceinline__ unsigned short f2bf(float f) {
  unsigned int u = __float_as_uint(f);
  u += 0x7FFFu + ((u >> 16) & 1u);   // round-to-nearest-even
  return (unsigned short)(u >> 16);
}

// ---------------- init / residual ----------------
__global__ void init_kernel(float* importance, int* counts, int* cursor) {
  int t = threadIdx.x;
  if (t < NEXP) { importance[t] = 0.f; counts[t] = 0; cursor[t] = 0; }
}

__global__ void residual_kernel(const float* __restrict__ x, float* __restrict__ out) {
  int i = blockIdx.x * blockDim.x + threadIdx.x;   // over N*D/4 float4
  ((float4*)out)[i] = ((const float4*)x)[i];
}

// ---------------- f32 -> bf16 batched transpose: dst[b][c][r] = src[b][r][c] ----
__global__ void transpose_bf16_kernel(const float* __restrict__ src,
                                      unsigned short* __restrict__ dst,
                                      int R, int C) {
  __shared__ unsigned short tile[32][33];
  size_t bofs = (size_t)blockIdx.z * R * C;
  int c0 = blockIdx.x * 32, r0 = blockIdx.y * 32;
  int tx = threadIdx.x, ty = threadIdx.y;          // 32 x 8
  #pragma unroll
  for (int i = 0; i < 32; i += 8) {
    tile[ty + i][tx] = f2bf(src[bofs + (size_t)(r0 + ty + i) * C + (c0 + tx)]);
  }
  __syncthreads();
  #pragma unroll
  for (int i = 0; i < 32; i += 8) {
    dst[bofs + (size_t)(c0 + ty + i) * R + (r0 + tx)] = tile[tx][ty + i];
  }
}

// ---------------- LayerNorm + router (one token per block) ----------------
__global__ __launch_bounds__(256)
void ln_gate_kernel(const float* __restrict__ x, const float* __restrict__ gw,
                    const float* __restrict__ lns, const float* __restrict__ lnb,
                    unsigned short* __restrict__ xn,
                    int* __restrict__ topi, float* __restrict__ topw,
                    float* __restrict__ importance, int* __restrict__ counts) {
  int n = blockIdx.x;
  int t = threadIdx.x;
  const float* xr = x + (size_t)n * DMODEL;
  float v[4];
  float s = 0.f, ss = 0.f;
  #pragma unroll
  for (int i = 0; i < 4; i++) {
    float f = xr[t + i * 256];
    v[i] = f; s += f; ss += f * f;
  }
  __shared__ float2 red[256];
  red[t] = make_float2(s, ss);
  __syncthreads();
  for (int st = 128; st > 0; st >>= 1) {
    if (t < st) { red[t].x += red[t + st].x; red[t].y += red[t + st].y; }
    __syncthreads();
  }
  float mu   = red[0].x * (1.f / DMODEL);
  float var  = red[0].y * (1.f / DMODEL) - mu * mu;
  float rstd = rsqrtf(var + 1e-5f);

  __shared__ float lg[NEXP];
  if (t < NEXP) lg[t] = 0.f;
  __syncthreads();

  float part[NEXP];
  #pragma unroll
  for (int e = 0; e < NEXP; e++) part[e] = 0.f;
  #pragma unroll
  for (int i = 0; i < 4; i++) {
    int d = t + i * 256;
    float xv = (v[i] - mu) * rstd * lns[d] + lnb[d];
    xn[(size_t)n * DMODEL + d] = f2bf(xv);
    #pragma unroll
    for (int e = 0; e < NEXP; e++) part[e] += xv * gw[d * NEXP + e];
  }
  #pragma unroll
  for (int e = 0; e < NEXP; e++) atomicAdd(&lg[e], part[e]);  // ds_add_f32
  __syncthreads();

  if (t == 0) {
    float p[NEXP];
    float mx = lg[0];
    #pragma unroll
    for (int e = 1; e < NEXP; e++) mx = fmaxf(mx, lg[e]);
    float sm = 0.f;
    #pragma unroll
    for (int e = 0; e < NEXP; e++) { p[e] = __expf(lg[e] - mx); sm += p[e]; }
    float inv = 1.f / sm;
    #pragma unroll
    for (int e = 0; e < NEXP; e++) { p[e] *= inv; atomicAdd(&importance[e], p[e]); }
    int i1 = 0;
    #pragma unroll
    for (int e = 1; e < NEXP; e++) if (p[e] > p[i1]) i1 = e;
    int i2 = (i1 == 0) ? 1 : 0;
    #pragma unroll
    for (int e = 0; e < NEXP; e++) if (e != i1 && p[e] > p[i2]) i2 = e;
    float sp = p[i1] + p[i2];
    topi[n * 2 + 0] = i1; topi[n * 2 + 1] = i2;
    topw[n * 2 + 0] = p[i1] / sp; topw[n * 2 + 1] = p[i2] / sp;
    atomicAdd(&counts[i1], 1);
    atomicAdd(&counts[i2], 1);
  }
}

// ---------------- scan offsets + aux loss ----------------
__global__ void scan_aux_kernel(const int* __restrict__ counts, int* __restrict__ offsets,
                                const float* __restrict__ importance, float* __restrict__ aux_out) {
  if (threadIdx.x == 0 && blockIdx.x == 0) {
    int acc = 0; float aux = 0.f;
    for (int e = 0; e < NEXP; e++) {
      offsets[e] = acc;
      acc += counts[e];
      aux += importance[e] * (float)counts[e];   // load[e] == counts[e]
    }
    offsets[NEXP] = acc;
    aux *= (float)NEXP / ((float)NTOK * (float)NTOK + 1e-6f);
    *aux_out = aux;
  }
}

// ---------------- routed slot fill ----------------
__global__ void fill_slots_kernel(const int* __restrict__ topi, const float* __restrict__ topw,
                                  const int* __restrict__ offsets, int* __restrict__ cursor,
                                  int* __restrict__ slot_tok, float* __restrict__ slot_w) {
  int n = blockIdx.x * blockDim.x + threadIdx.x;
  if (n >= NTOK) return;
  #pragma unroll
  for (int k = 0; k < KTOP; k++) {
    int e = topi[n * 2 + k];
    int pos = atomicAdd(&cursor[e], 1);
    int s = offsets[e] + pos;
    slot_tok[s] = n;
    slot_w[s]   = topw[n * 2 + k];
  }
}

// async global->LDS copy of 32 bytes (two b128 transfers), gfx1250 path.
// The INST_OFFSET is added to BOTH the LDS destination and the global address
// (CDNA5 ISA), so one offset:16 advances both sides.
__device__ __forceinline__ void async_copy_b256(unsigned lds_byte_off,
                                                const void* gptr) {
  unsigned long long ga = (unsigned long long)(uintptr_t)gptr;
  asm volatile(
      "global_load_async_to_lds_b128 %0, %1, off\n\t"
      "global_load_async_to_lds_b128 %0, %1, off offset:16"
      :: "v"(lds_byte_off), "v"(ga));
}

__device__ __forceinline__ void async_wait_all() {
  asm volatile("s_wait_asynccnt 0x0" ::: "memory");
}

// ---------------- GEMM1: H = gelu(Xn_routed @ W1[e] + b1[e]) ----------------
__global__ __launch_bounds__(256)
void gemm1_kernel(const unsigned short* __restrict__ xn,    // [N, D] bf16
                  const unsigned short* __restrict__ w1t,   // [E, F, D] bf16 (transposed)
                  const float* __restrict__ b1,             // [E, F]
                  const int* __restrict__ counts, const int* __restrict__ offsets,
                  const int* __restrict__ slot_tok,
                  unsigned short* __restrict__ H) {         // [SLOTS, F] bf16
  __shared__ unsigned short bsh[TN][KS];                    // 8 KB B tile
  // map block -> (expert, m-tile)
  int b = blockIdx.x, e = -1, mt = 0, tbase = 0, cnt = 0, off = 0;
  for (int i = 0; i < NEXP; i++) {
    int ci = counts[i];
    int ti = (ci + TM - 1) / TM;
    if (b < tbase + ti) { e = i; mt = b - tbase; cnt = ci; off = offsets[i]; break; }
    tbase += ti;
  }
  if (e < 0) return;
  int n0   = blockIdx.y * TN;
  int tid  = threadIdx.x;
  int lane = tid & 31;
  int wave = tid >> 5;
  int wm   = wave >> 1;                 // 0..3 (M groups of 16)
  int wn   = wave & 1;                  // 0..1 (N groups of 64)
  int m0   = mt * TM;

  int arow = m0 + wm * 16 + (lane & 15);
  int atok = slot_tok[off + ((arow < cnt) ? arow : (cnt - 1))];
  const unsigned short* aptr = xn + (size_t)atok * DMODEL;
  int koffA = (lane >> 4) * 8;          // bf16 A layout per lane
  int krowB = (lane >> 4) * 8;          // bf16 B layout per lane

  f32x8 acc[4];
  f32x8 zero = {0.f, 0.f, 0.f, 0.f, 0.f, 0.f, 0.f, 0.f};
  #pragma unroll
  for (int j = 0; j < 4; j++) acc[j] = zero;

  int bc = tid >> 1;                    // 0..127 col
  int bk = (tid & 1) * 16;              // 0/16 K half (16 bf16 = 32 B)
  const unsigned short* bsrc = w1t + ((size_t)e * DFF + n0 + bc) * DMODEL + bk;
  unsigned lds_dst = (unsigned)(uintptr_t)&bsh[bc][bk];     // LDS byte offset

  for (int k0 = 0; k0 < DMODEL; k0 += KS) {
    // async global -> LDS staging of the 128x32 B tile (no VGPR round-trip)
    async_copy_b256(lds_dst, bsrc + k0);

    BFrag a;                            // overlaps with the async copy
    a.u[0] = *(const uint4*)(aptr + k0 + koffA);
    a.u[1] = *(const uint4*)(aptr + k0 + koffA + 16);
    if (k0 + KS < DMODEL) __builtin_prefetch(aptr + k0 + KS + koffA, 0, 0);

    async_wait_all();
    __syncthreads();

    #pragma unroll
    for (int j = 0; j < 4; j++) {
      BFrag bb;
      const unsigned short* lp = &bsh[wn * 64 + j * 16 + (lane & 15)][krowB];
      bb.u[0] = *(const uint4*)(lp);
      bb.u[1] = *(const uint4*)(lp + 16);
      acc[j] = __builtin_amdgcn_wmma_f32_16x16x32_bf16(
          false, a.v, false, bb.v, (short)0, acc[j], false, false);
    }
    __syncthreads();
  }

  int mBase = wm * 16 + (lane >> 4) * 8;
  int col0  = n0 + wn * 64;
  #pragma unroll
  for (int j = 0; j < 4; j++) {
    int f = col0 + j * 16 + (lane & 15);
    float bias = b1[e * DFF + f];
    #pragma unroll
    for (int r = 0; r < 8; r++) {
      int row = m0 + mBase + r;
      if (row < cnt) {
        float vv = acc[j][r] + bias;
        vv = 0.5f * vv * (1.0f + erff(vv * 0.70710678118654752f));   // exact GELU
        H[(size_t)(off + row) * DFF + f] = f2bf(vv);
      }
    }
  }
}

// ---------------- GEMM2: out += combine * (H @ W2[e] + b2[e]) ----------------
__global__ __launch_bounds__(256)
void gemm2_kernel(const unsigned short* __restrict__ H,     // [SLOTS, F] bf16
                  const unsigned short* __restrict__ w2t,   // [E, D, F] bf16 (transposed)
                  const float* __restrict__ b2,             // [E, D]
                  const int* __restrict__ counts, const int* __restrict__ offsets,
                  const int* __restrict__ slot_tok, const float* __restrict__ slot_w,
                  float* __restrict__ out) {                // [N, D], residual-initialized
  __shared__ unsigned short bsh[TN][KS];
  int b = blockIdx.x, e = -1, mt = 0, tbase = 0, cnt = 0, off = 0;
  for (int i = 0; i < NEXP; i++) {
    int ci = counts[i];
    int ti = (ci + TM - 1) / TM;
    if (b < tbase + ti) { e = i; mt = b - tbase; cnt = ci; off = offsets[i]; break; }
    tbase += ti;
  }
  if (e < 0) return;
  int n0   = blockIdx.y * TN;
  int tid  = threadIdx.x;
  int lane = tid & 31;
  int wave = tid >> 5;
  int wm   = wave >> 1;
  int wn   = wave & 1;
  int m0   = mt * TM;

  int arow  = m0 + wm * 16 + (lane & 15);
  int aslot = off + ((arow < cnt) ? arow : (cnt - 1));
  const unsigned short* aptr = H + (size_t)aslot * DFF;
  int koffA = (lane >> 4) * 8;
  int krowB = (lane >> 4) * 8;

  f32x8 acc[4];
  f32x8 zero = {0.f, 0.f, 0.f, 0.f, 0.f, 0.f, 0.f, 0.f};
  #pragma unroll
  for (int j = 0; j < 4; j++) acc[j] = zero;

  int bc = tid >> 1;
  int bk = (tid & 1) * 16;
  const unsigned short* bsrc = w2t + ((size_t)e * DMODEL + n0 + bc) * DFF + bk;
  unsigned lds_dst = (unsigned)(uintptr_t)&bsh[bc][bk];

  for (int k0 = 0; k0 < DFF; k0 += KS) {
    async_copy_b256(lds_dst, bsrc + k0);

    BFrag a;
    a.u[0] = *(const uint4*)(aptr + k0 + koffA);
    a.u[1] = *(const uint4*)(aptr + k0 + koffA + 16);
    if (k0 + KS < DFF) __builtin_prefetch(aptr + k0 + KS + koffA, 0, 0);

    async_wait_all();
    __syncthreads();

    #pragma unroll
    for (int j = 0; j < 4; j++) {
      BFrag bb;
      const unsigned short* lp = &bsh[wn * 64 + j * 16 + (lane & 15)][krowB];
      bb.u[0] = *(const uint4*)(lp);
      bb.u[1] = *(const uint4*)(lp + 16);
      acc[j] = __builtin_amdgcn_wmma_f32_16x16x32_bf16(
          false, a.v, false, bb.v, (short)0, acc[j], false, false);
    }
    __syncthreads();
  }

  int mBase = wm * 16 + (lane >> 4) * 8;
  int col0  = n0 + wn * 64;
  #pragma unroll
  for (int j = 0; j < 4; j++) {
    int d = col0 + j * 16 + (lane & 15);
    float bias = b2[e * DMODEL + d];
    #pragma unroll
    for (int r = 0; r < 8; r++) {
      int row = m0 + mBase + r;
      if (row < cnt) {
        int s   = off + row;
        int tok = slot_tok[s];
        float w = slot_w[s];
        atomicAdd(&out[(size_t)tok * DMODEL + d], (acc[j][r] + bias) * w);
      }
    }
  }
}

// ---------------- host launcher ----------------
extern "C" void kernel_launch(void* const* d_in, const int* in_sizes, int n_in,
                              void* d_out, int out_size, void* d_ws, size_t ws_size,
                              hipStream_t stream) {
  const float* x      = (const float*)d_in[0];  // [N, D]
  const float* gate_w = (const float*)d_in[1];  // [D, E]
  const float* ln_s   = (const float*)d_in[2];  // [D]
  const float* ln_b   = (const float*)d_in[3];  // [D]
  const float* w1     = (const float*)d_in[4];  // [E, D, F]
  const float* b1     = (const float*)d_in[5];  // [E, F]
  const float* w2     = (const float*)d_in[6];  // [E, F, D]
  const float* b2     = (const float*)d_in[7];  // [E, D]
  float* out = (float*)d_out;                   // [N*D] out, then 1 aux scalar

  char* ws = (char*)d_ws;
  size_t o = 0;
  auto alloc = [&](size_t bytes) { size_t r = o; o = (o + bytes + 255) & ~(size_t)255; return r; };
  unsigned short* xn_bf    = (unsigned short*)(ws + alloc((size_t)NTOK * DMODEL * 2));
  unsigned short* w1t      = (unsigned short*)(ws + alloc((size_t)NEXP * DFF * DMODEL * 2));
  unsigned short* w2t      = (unsigned short*)(ws + alloc((size_t)NEXP * DMODEL * DFF * 2));
  unsigned short* Hbuf     = (unsigned short*)(ws + alloc((size_t)SLOTS * DFF * 2));
  int*   topi      = (int*)  (ws + alloc((size_t)NTOK * 2 * 4));
  float* topw      = (float*)(ws + alloc((size_t)NTOK * 2 * 4));
  int*   slot_tok  = (int*)  (ws + alloc((size_t)SLOTS * 4));
  float* slot_w    = (float*)(ws + alloc((size_t)SLOTS * 4));
  int*   counts    = (int*)  (ws + alloc(NEXP * 4));
  int*   offsets   = (int*)  (ws + alloc((NEXP + 1) * 4));
  int*   cursor    = (int*)  (ws + alloc(NEXP * 4));
  float* importance= (float*)(ws + alloc(NEXP * 4));

  init_kernel<<<1, 32, 0, stream>>>(importance, counts, cursor);
  residual_kernel<<<(NTOK * DMODEL / 4) / 256, 256, 0, stream>>>(x, out);
  // W1 [E,D,F] -> W1T [E,F,D]
  transpose_bf16_kernel<<<dim3(DFF / 32, DMODEL / 32, NEXP), dim3(32, 8), 0, stream>>>(w1, w1t, DMODEL, DFF);
  // W2 [E,F,D] -> W2T [E,D,F]
  transpose_bf16_kernel<<<dim3(DMODEL / 32, DFF / 32, NEXP), dim3(32, 8), 0, stream>>>(w2, w2t, DFF, DMODEL);
  ln_gate_kernel<<<NTOK, 256, 0, stream>>>(x, gate_w, ln_s, ln_b, xn_bf, topi, topw, importance, counts);
  scan_aux_kernel<<<1, 32, 0, stream>>>(counts, offsets, importance, out + (size_t)NTOK * DMODEL);
  fill_slots_kernel<<<NTOK / 256, 256, 0, stream>>>(topi, topw, offsets, cursor, slot_tok, slot_w);
  gemm1_kernel<<<dim3(MAX_TILES, DFF / TN), 256, 0, stream>>>(xn_bf, w1t, b1, counts, offsets, slot_tok, Hbuf);
  gemm2_kernel<<<dim3(MAX_TILES, DMODEL / TN), 256, 0, stream>>>(Hbuf, w2t, b2, counts, offsets, slot_tok, slot_w, out);
}
